// MLP_decoder_right_41283225649498
// MI455X (gfx1250) — matrix operand order repaired and checked by reading, not compiled, not run on hardware
//
#include <hip/hip_runtime.h>
#include <hip/hip_bf16.h>

// ---------------------------------------------------------------------------
// Problem: 500K pairs through a [512]->[512]->[256]->[1] leaky-ReLU MLP,
// scattered into a 20000x5000 zeroed output.
// MI455X plan: bf16 WMMA (v_wmma_f32_16x16x32_bf16), activations in LDS,
// weights pre-converted to bf16 once into d_ws (768KB -> L2 resident).
// ---------------------------------------------------------------------------

typedef __attribute__((ext_vector_type(16))) __bf16 v16bf;
typedef __attribute__((ext_vector_type(8)))  __bf16 v8bf;
typedef __attribute__((ext_vector_type(8)))  float  v8f;

#define DIN     512   // IN1 + IN2
#define H1      512   // DIM_PRED
#define H2      256   // DIM_PRED/2
#define NPROT   5000
#define TILE_M  16
#define SA      520   // LDS row stride (bf16 elems), 16B-aligned rows, conflict pad
#define SB      520
#define SC      264

// --- A fragment (16x32 bf16, M x K) from LDS, contiguous k order, same as B ----------------
// lane 0-15 : M=lane,    K = kb+0..15
// lane 16-31: M=lane-16, K = kb+16..31
__device__ __forceinline__ v16bf load_a_frag(const __bf16* lds, int stride,
                                             int lane, int kb) {
    const int m = lane & 15;
    const int s = (lane < 16) ? 0 : 16;
    const __bf16* p = lds + m * stride + kb + s;
    v8bf lo = *(const v8bf*)(p);
    v8bf hi = *(const v8bf*)(p + 8);
    return __builtin_shufflevector(lo, hi, 0,1,2,3,4,5,6,7,8,9,10,11,12,13,14,15);
}

// --- B fragment (32x16 bf16, K x N) where B[k][n] = W[n][k] -----------------
// lane 0-15 : N=lane,    K = kb+0..15   (16 contiguous bf16 of weight row n)
// lane 16-31: N=lane-16, K = kb+16..31
__device__ __forceinline__ v16bf load_b_frag(const __bf16* __restrict__ W, int ldw,
                                             int nbase, int lane, int kb) {
    const int n    = nbase + (lane & 15);
    const int koff = (lane < 16) ? 0 : 16;
    return *(const v16bf*)(W + (size_t)n * ldw + kb + koff);
}

__device__ __forceinline__ float leaky(float x) { return x > 0.f ? x : 0.01f * x; }

// ---------------------------------------------------------------------------
__global__ void convert_weights_kernel(const float* __restrict__ W1,
                                       const float* __restrict__ W2,
                                       __bf16* __restrict__ ws) {
    const int n1 = H1 * DIN;       // 262144
    const int n2 = H2 * H1;        // 131072
    int i = blockIdx.x * blockDim.x + threadIdx.x;
    if (i < n1)            ws[i] = (__bf16)W1[i];
    else if (i < n1 + n2)  ws[i] = (__bf16)W2[i - n1];
}

__global__ void zero_out_kernel(float4* __restrict__ out, long long n4) {
    long long i      = (long long)blockIdx.x * blockDim.x + threadIdx.x;
    long long stride = (long long)gridDim.x * blockDim.x;
    float4 z; z.x = z.y = z.z = z.w = 0.f;
    for (; i < n4; i += stride) out[i] = z;
}

__global__ void __launch_bounds__(256)
mlp_pairs_kernel(const float* __restrict__ x0, const float* __restrict__ x1,
                 const int* __restrict__ idx0, const int* __restrict__ idx1,
                 const __bf16* __restrict__ W1bf, const float* __restrict__ b1,
                 const __bf16* __restrict__ W2bf, const float* __restrict__ b2,
                 const float* __restrict__ W3,   const float* __restrict__ b3,
                 float* __restrict__ out) {
    __shared__ __bf16 actA[TILE_M * SA];   // gathered input  [16 x 512]
    __shared__ __bf16 actB[TILE_M * SB];   // hidden1         [16 x 512]
    __shared__ __bf16 actC[TILE_M * SC];   // hidden2         [16 x 256]

    const int tid  = threadIdx.x;
    const int lane = tid & 31;
    const int wave = tid >> 5;              // 8 wave32s
    const long long pbase = (long long)blockIdx.x * TILE_M;

    // ---- Stage 0: gather 16 pairs, concat(x0row, x1row), fp32 -> bf16 ------
    // 16 pairs * 128 float4 = 2048 chunks; 8 per thread.
    for (int it = tid; it < TILE_M * 128; it += 256) {
        const int m = it >> 7;
        const int q = it & 127;
        const long long p = pbase + m;
        const float* src = (q < 64)
            ? (x0 + (long long)idx0[p] * 256 + q * 4)
            : (x1 + (long long)idx1[p] * 256 + (q - 64) * 4);
        float4 v = *(const float4*)src;
        __bf16* dst = &actA[m * SA + q * 4];
        dst[0] = (__bf16)v.x; dst[1] = (__bf16)v.y;
        dst[2] = (__bf16)v.z; dst[3] = (__bf16)v.w;
    }
    __syncthreads();

    // ---- Layer 1: h1 = leaky(actA @ W1^T + b1)  -> actB [16 x 512] ---------
    // 32 N-tiles / 8 waves = 4 tiles per wave; K = 512 = 16 WMMA steps.
    for (int nt = wave * 4; nt < wave * 4 + 4; ++nt) {
        v8f c = {};
        #pragma unroll
        for (int kb = 0; kb < DIN; kb += 32) {
            v16bf a = load_a_frag(actA, SA, lane, kb);
            v16bf b = load_b_frag(W1bf, DIN, nt * 16, lane, kb);
            c = __builtin_amdgcn_wmma_f32_16x16x32_bf16(
                    false, a, false, b, (short)0, c, false, false);
        }
        const int n    = nt * 16 + (lane & 15);
        const float bn = b1[n];
        const int mof  = (lane < 16) ? 0 : 8;    // C/D layout: VGPR r -> M=r(+8)
        #pragma unroll
        for (int r = 0; r < 8; ++r)
            actB[(r + mof) * SB + n] = (__bf16)leaky(c[r] + bn);
    }
    __syncthreads();

    // ---- Layer 2: h2 = leaky(actB @ W2^T + b2)  -> actC [16 x 256] ---------
    for (int nt = wave * 2; nt < wave * 2 + 2; ++nt) {
        v8f c = {};
        #pragma unroll
        for (int kb = 0; kb < H1; kb += 32) {
            v16bf a = load_a_frag(actB, SB, lane, kb);
            v16bf b = load_b_frag(W2bf, H1, nt * 16, lane, kb);
            c = __builtin_amdgcn_wmma_f32_16x16x32_bf16(
                    false, a, false, b, (short)0, c, false, false);
        }
        const int n    = nt * 16 + (lane & 15);
        const float bn = b2[n];
        const int mof  = (lane < 16) ? 0 : 8;
        #pragma unroll
        for (int r = 0; r < 8; ++r)
            actC[(r + mof) * SC + n] = (__bf16)leaky(c[r] + bn);
    }
    __syncthreads();

    // ---- Layer 3: o[m] = leaky(dot(h2[m], W3) + b3); scatter ---------------
    // 16 lanes per pair, 16 elems per lane, shuffle reduce within wave.
    {
        const int m = tid >> 4;
        const int j = tid & 15;
        float s = 0.f;
        #pragma unroll
        for (int i = 0; i < 16; ++i) {
            const int k = j * 16 + i;
            s += (float)actC[m * SC + k] * W3[k];
        }
        #pragma unroll
        for (int off = 8; off >= 1; off >>= 1)
            s += __shfl_xor(s, off, 16);
        if (j == 0) {
            const float o = leaky(s + b3[0]);
            const long long p = pbase + m;
            out[(long long)idx0[p] * NPROT + idx1[p]] = o;
        }
    }
}

// ---------------------------------------------------------------------------
extern "C" void kernel_launch(void* const* d_in, const int* in_sizes, int n_in,
                              void* d_out, int out_size, void* d_ws, size_t ws_size,
                              hipStream_t stream) {
    const float* x0   = (const float*)d_in[0];
    const float* x1   = (const float*)d_in[1];
    const int*   idx0 = (const int*)d_in[2];
    const int*   idx1 = (const int*)d_in[3];
    const float* W1   = (const float*)d_in[4];
    const float* b1   = (const float*)d_in[5];
    const float* W2   = (const float*)d_in[6];
    const float* b2   = (const float*)d_in[7];
    const float* W3   = (const float*)d_in[8];
    const float* b3   = (const float*)d_in[9];
    float* out = (float*)d_out;

    __bf16* W1bf = (__bf16*)d_ws;            // 512*512 bf16
    __bf16* W2bf = W1bf + H1 * DIN;          // 256*512 bf16 (total 768 KB)

    const int n_pairs = in_sizes[2];         // 500000, divisible by 16
    const int nconv   = H1 * DIN + H2 * H1;

    convert_weights_kernel<<<(nconv + 255) / 256, 256, 0, stream>>>(W1, W2, W1bf);
    zero_out_kernel<<<8192, 256, 0, stream>>>((float4*)out, (long long)out_size / 4);
    mlp_pairs_kernel<<<n_pairs / TILE_M, 256, 0, stream>>>(
        x0, x1, idx0, idx1, W1bf, b1, W2bf, b2, W3, b3, out);
}
